// MetricModel_30889404793008
// MI455X (gfx1250) — matrix operand order repaired and checked by reading, not compiled, not run on hardware
//
#include <hip/hip_runtime.h>
#include <hip/hip_bf16.h>
#include <math.h>

// ---------------------------------------------------------------------------
// Problem constants (from reference)
// ---------------------------------------------------------------------------
#define N_WAY   64
#define K_SHOT  5
#define Q_PER   50
#define R_TOT   (N_WAY * (K_SHOT + Q_PER))   // 3520 rows of x
#define NQ      (N_WAY * Q_PER)              // 3200 queries
#define D_INP   8192
#define D_FEAT  2048
#define TOPK    10

typedef __attribute__((ext_vector_type(16))) __bf16 v16bf;
typedef __attribute__((ext_vector_type(8)))  __bf16 v8bf;
typedef __attribute__((ext_vector_type(4)))  __bf16 v4bf;
typedef __attribute__((ext_vector_type(8)))  float  v8f;
typedef __attribute__((ext_vector_type(4)))  int    v4i;

// Address-space-qualified vector pointers for the async-LDS builtin
// (diagnostic showed param 0 is "v4i __device__ *", i.e. v4i addrspace(1)*).
typedef __attribute__((address_space(1))) v4i v4i_global;
typedef __attribute__((address_space(3))) v4i v4i_lds;

// ---------------------------------------------------------------------------
// CDNA5 async Global->LDS staging (ASYNCcnt path), with portable fallback.
// Each lane moves one 16-byte chunk: global bf16 x8 -> LDS bf16 x8.
// ---------------------------------------------------------------------------
#if __has_builtin(__builtin_amdgcn_global_load_async_to_lds_b128)
#define ASYNC_LDS 1
#endif

__device__ __forceinline__ void stage16(const __bf16* g, __bf16* l) {
#ifdef ASYNC_LDS
    __builtin_amdgcn_global_load_async_to_lds_b128(
        (v4i_global*)g, (v4i_lds*)l, 0, 0);
#else
    *(v8bf*)l = *(const v8bf*)g;
#endif
}

__device__ __forceinline__ void wait_stage() {
#ifdef ASYNC_LDS
#if __has_builtin(__builtin_amdgcn_s_wait_asynccnt)
    __builtin_amdgcn_s_wait_asynccnt(0);
#else
    asm volatile("s_wait_asynccnt 0x0" ::: "memory");
#endif
#endif
}

// ---------------------------------------------------------------------------
// WMMA fragment loaders (per CDNA5 ISA 7.12.2 VGPR layouts, wave32)
//   A 16x32 bf16: lanes 0-15 -> M=lane,  K groups {0-7,16-23}; lanes 16-31 +8
//   B 32x16 bf16 (tile stored transposed in LDS as [N][32]): N = lane&15,
//     K groups {0-7,16-23}; lanes 16-31 +8 (same as A)
// ---------------------------------------------------------------------------
__device__ __forceinline__ v16bf frag_a(const __bf16* As /*[64][32]*/, int tm, int lane) {
    const __bf16* row = As + (tm + (lane & 15)) * 32 + ((lane >> 4) << 3); // +0 or +8
    v16bf a;
#pragma unroll
    for (int i = 0; i < 8; ++i) a[i] = row[i];          // K = base + 0..7
#pragma unroll
    for (int i = 0; i < 8; ++i) a[8 + i] = row[16 + i]; // K = base + 16..23
    return a;
}

__device__ __forceinline__ v16bf frag_b(const __bf16* Bst /*[64][32]*/, int tn, int lane) {
    const __bf16* row = Bst + (tn + (lane & 15)) * 32 + ((lane >> 4) << 3); // +0 or +8
    v16bf b;
#pragma unroll
    for (int i = 0; i < 8; ++i) { b[i] = row[i]; b[8 + i] = row[16 + i]; }
    return b;
}

// ---------------------------------------------------------------------------
// bf16 NT GEMM: C[M,N] = A[M,K] * B[N,K]^T   (both bf16 row-major)
// Optional fused epilogue: C = scale * (2*acc - rnA[m] - rnB[n]) (neg-sq-euclid)
// Block tile 64x64, K-step 32, 256 threads = 8 waves, double-buffered LDS,
// async Global->LDS staging overlapped with WMMA issue.
// All dims are multiples of 64/32 -> no guards, EXEC stays all-ones.
// ---------------------------------------------------------------------------
__global__ void __launch_bounds__(256)
gemm64_bf16nt(const __bf16* __restrict__ A, const __bf16* __restrict__ B,
              float* __restrict__ C, int M, int N, int K,
              const float* __restrict__ rnA, const float* __restrict__ rnB,
              const float* __restrict__ scale_ptr)
{
    __shared__ __bf16 As[2][64 * 32];  // [m][k]
    __shared__ __bf16 Bs[2][64 * 32];  // [n][k]

    const int tid  = threadIdx.x;
    const int lane = tid & 31;
    const int w    = tid >> 5;
    const int m0   = blockIdx.y * 64;
    const int n0   = blockIdx.x * 64;
    const int tm   = (w & 3) * 16;        // wave's M tile within block
    const int tn   = (w >> 2) * 32;       // wave's 32-wide N strip

    // One 16B chunk per lane: row r = tid/4, col chunk c8 = (tid%4)*8.
    // LDS dest offset r*32 + c8 == tid*8 elements (16B per lane).
    const __bf16* aSrc = A + (size_t)(m0 + (tid >> 2)) * K + ((tid & 3) << 3);
    const __bf16* bSrc = B + (size_t)(n0 + (tid >> 2)) * K + ((tid & 3) << 3);

    v8f acc0 = {};
    v8f acc1 = {};

    stage16(aSrc, &As[0][tid * 8]);
    stage16(bSrc, &Bs[0][tid * 8]);
    wait_stage();
    __syncthreads();

    int cur = 0;
    for (int k0 = 0; k0 < K; k0 += 32) {
        const int nxt = k0 + 32;
        if (nxt < K) {                       // prefetch next tiles into other buffer
            stage16(aSrc + nxt, &As[cur ^ 1][tid * 8]);
            stage16(bSrc + nxt, &Bs[cur ^ 1][tid * 8]);
        }
        v16bf a  = frag_a(&As[cur][0], tm, lane);
        v16bf b0 = frag_b(&Bs[cur][0], tn, lane);
        v16bf b1 = frag_b(&Bs[cur][0], tn + 16, lane);
        acc0 = __builtin_amdgcn_wmma_f32_16x16x32_bf16(false, a, false, b0,
                                                       (short)0, acc0, false, false);
        acc1 = __builtin_amdgcn_wmma_f32_16x16x32_bf16(false, a, false, b1,
                                                       (short)0, acc1, false, false);
        if (nxt < K) {
            wait_stage();                    // overlap: waited after WMMA issue
            __syncthreads();
            cur ^= 1;
        }
    }

    // ---- epilogue (C/D layout: VGPR v -> M = v + 8*(lane>=16), N = lane&15) ----
    const float s   = scale_ptr ? scale_ptr[0] : 1.0f;
    const int mbase = m0 + tm + ((lane >> 4) << 3);
    const int nc0   = n0 + tn + (lane & 15);
    const int nc1   = nc0 + 16;
#pragma unroll
    for (int v = 0; v < 8; ++v) {
        int m = mbase + v;
        float r0 = acc0[v], r1 = acc1[v];
        if (rnA) {
            float ra = rnA[m];
            r0 = 2.0f * r0 - ra - rnB[nc0];
            r1 = 2.0f * r1 - ra - rnB[nc1];
        }
        C[(size_t)m * N + nc0] = r0 * s;
        C[(size_t)m * N + nc1] = r1 * s;
    }
}

// ---------------------------------------------------------------------------
// One-time precision conversion kernels (f32 -> bf16), vectorized 4/thread
// ---------------------------------------------------------------------------
__global__ void cvt_bf16_kernel(const float* __restrict__ src, __bf16* __restrict__ dst) {
    size_t i = ((size_t)blockIdx.x * 256 + threadIdx.x) * 4;
    float4 v = *(const float4*)(src + i);
    v4bf b;
    b[0] = (__bf16)v.x; b[1] = (__bf16)v.y; b[2] = (__bf16)v.z; b[3] = (__bf16)v.w;
    *(v4bf*)(dst + i) = b;
}

// W [8192,2048] f32 -> Wt [2048,8192] bf16 (so the big GEMM is NT)
__global__ void transpose_cvt_kernel(const float* __restrict__ W, __bf16* __restrict__ Wt) {
    __shared__ float tile[64][65];
    int n0 = blockIdx.x * 64;
    int k0 = blockIdx.y * 64;
    int t = threadIdx.x;
#pragma unroll
    for (int e = 0; e < 16; ++e) {
        int l = e * 256 + t;
        int r = l >> 6, c = l & 63;                       // r = k, c = n (coalesced)
        tile[r][c] = W[(size_t)(k0 + r) * D_FEAT + (n0 + c)];
    }
    __syncthreads();
#pragma unroll
    for (int e = 0; e < 16; ++e) {
        int l = e * 256 + t;
        int r = l >> 6, c = l & 63;                       // r = n, c = k (coalesced)
        Wt[(size_t)(n0 + r) * D_INP + (k0 + c)] = (__bf16)tile[c][r];
    }
}

// ---------------------------------------------------------------------------
// Glue kernels (tiny traffic vs. the GEMMs)
// ---------------------------------------------------------------------------
__global__ void proto_kernel(const float* __restrict__ feat, float* __restrict__ proto) {
    int c = blockIdx.x;
    int d = blockIdx.y * 256 + threadIdx.x;
    float s = 0.f;
#pragma unroll
    for (int k = 0; k < K_SHOT; ++k)
        s += feat[(size_t)(c * (K_SHOT + Q_PER) + k) * D_FEAT + d];
    proto[(size_t)c * D_FEAT + d] = s * (1.0f / K_SHOT);
}

__global__ void query_copy_kernel(const float* __restrict__ feat, float* __restrict__ query) {
    int i = blockIdx.x;
    int d = blockIdx.y * 256 + threadIdx.x;
    int cls = i / Q_PER, r = i % Q_PER;
    query[(size_t)i * D_FEAT + d] =
        feat[(size_t)(cls * (K_SHOT + Q_PER) + K_SHOT + r) * D_FEAT + d];
}

__global__ void row_norm_kernel(const float* __restrict__ X, float* __restrict__ out, int D) {
    __shared__ float red[256];
    int i = blockIdx.x;
    float s = 0.f;
    for (int d = threadIdx.x; d < D; d += 256) {
        float v = X[(size_t)i * D + d];
        s += v * v;
    }
    red[threadIdx.x] = s;
    __syncthreads();
    for (int off = 128; off > 0; off >>= 1) {
        if (threadIdx.x < off) red[threadIdx.x] += red[threadIdx.x + off];
        __syncthreads();
    }
    if (threadIdx.x == 0) out[i] = red[0];
}

__global__ void argmax_kernel(const float* __restrict__ pre_sim, int* __restrict__ label) {
    int i = blockIdx.x * 256 + threadIdx.x;
    if (i >= NQ) return;
    const float* row = pre_sim + (size_t)i * N_WAY;
    float best = row[0];
    int bi = 0;
    for (int c = 1; c < N_WAY; ++c) {
        float v = row[c];
        if (v > best) { best = v; bi = c; }   // strict > : first max wins (jax tie rule)
    }
    label[i] = bi;
}

// Per-class softmax stats over assigned queries; extra logit 0 appended.
__global__ void cls_stats_kernel(const float* __restrict__ pre_sim, const int* __restrict__ label,
                                 float* __restrict__ mArr, float* __restrict__ ZArr,
                                 float* __restrict__ pcoef) {
    __shared__ float red[256];
    __shared__ float mSh;
    int c = blockIdx.x, t = threadIdx.x;
    float lm = -3.0e38f;
    for (int j = t; j < NQ; j += 256)
        if (label[j] == c) lm = fmaxf(lm, pre_sim[(size_t)j * N_WAY + c]);
    red[t] = lm;
    __syncthreads();
    for (int off = 128; off > 0; off >>= 1) {
        if (t < off) red[t] = fmaxf(red[t], red[t + off]);
        __syncthreads();
    }
    if (t == 0) mSh = fmaxf(red[0], 0.0f);   // include zero column
    __syncthreads();
    float m = mSh, lz = 0.f;
    for (int j = t; j < NQ; j += 256)
        if (label[j] == c) lz += expf(pre_sim[(size_t)j * N_WAY + c] - m);
    red[t] = lz;
    __syncthreads();
    for (int off = 128; off > 0; off >>= 1) {
        if (t < off) red[t] += red[t + off];
        __syncthreads();
    }
    if (t == 0) {
        float Z = red[0] + expf(0.0f - m);
        mArr[c] = m; ZArr[c] = Z; pcoef[c] = expf(0.0f - m) / Z;
    }
}

__global__ void coeff_kernel(const float* __restrict__ pre_sim, const int* __restrict__ label,
                             const float* __restrict__ mArr, const float* __restrict__ ZArr,
                             float* __restrict__ coeff) {
    int i = blockIdx.x * 256 + threadIdx.x;
    if (i >= NQ) return;
    int c = label[i];
    coeff[i] = expf(pre_sim[(size_t)i * N_WAY + c] - mArr[c]) / ZArr[c];
}

// adapted_proto[c] = sum_{j: label j==c} coeff[j]*query[j] + pcoef[c]*proto[c]
__global__ void adapted_proto_kernel(const float* __restrict__ query, const float* __restrict__ proto,
                                     const int* __restrict__ label, const float* __restrict__ coeff,
                                     const float* __restrict__ pcoef, float* __restrict__ aproto) {
    int c = blockIdx.x, t = threadIdx.x;
    float acc[8];
#pragma unroll
    for (int s = 0; s < 8; ++s) acc[s] = 0.f;
    for (int j = 0; j < NQ; ++j) {
        if (label[j] == c) {                 // uniform across block
            float wj = coeff[j];
            const float* qr = query + (size_t)j * D_FEAT;
#pragma unroll
            for (int s = 0; s < 8; ++s) acc[s] += wj * qr[t + s * 256];
        }
    }
    float pc = pcoef[c];
#pragma unroll
    for (int s = 0; s < 8; ++s) {
        int d = t + s * 256;
        aproto[(size_t)c * D_FEAT + d] = acc[s] + pc * proto[(size_t)c * D_FEAT + d];
    }
}

// Top-10 per row (insertion sort; strict > keeps lower index on ties like lax.top_k)
__global__ void topk_kernel(const float* __restrict__ qsim, int* __restrict__ kind) {
    int i = blockIdx.x * 64 + threadIdx.x;
    if (i >= NQ) return;
    float v[TOPK]; int id[TOPK];
#pragma unroll
    for (int t = 0; t < TOPK; ++t) { v[t] = -3.0e38f; id[t] = -1; }
    const float* row = qsim + (size_t)i * NQ;
    for (int j = 0; j < NQ; ++j) {
        float s = row[j];
        if (s > v[TOPK - 1]) {
            int p = TOPK - 1;
            while (p > 0 && s > v[p - 1]) { v[p] = v[p - 1]; id[p] = id[p - 1]; --p; }
            v[p] = s; id[p] = j;
        }
    }
#pragma unroll
    for (int t = 0; t < TOPK; ++t) kind[(size_t)i * TOPK + t] = id[t];
}

// Mutual-kNN softmax + weighted sum (<=10 nonzeros per row; self is always mutual)
__global__ void adapted_query_kernel(const float* __restrict__ qsim, const int* __restrict__ kind,
                                     const float* __restrict__ query, float* __restrict__ aq) {
    __shared__ int   cIdx[TOPK];
    __shared__ float cW[TOPK];
    __shared__ int   cCnt;
    int i = blockIdx.x;
    if (threadIdx.x == 0) {
        int idx[TOPK]; float vals[TOPK]; int cnt = 0;
        for (int t = 0; t < TOPK; ++t) {
            int j = kind[(size_t)i * TOPK + t];
            bool mut = false;
            for (int u = 0; u < TOPK; ++u)
                if (kind[(size_t)j * TOPK + u] == i) mut = true;
            if (mut) { idx[cnt] = j; vals[cnt] = qsim[(size_t)i * NQ + j]; ++cnt; }
        }
        float m = -3.0e38f;
        for (int t = 0; t < cnt; ++t) m = fmaxf(m, vals[t]);
        float Z = 0.f;
        for (int t = 0; t < cnt; ++t) Z += expf(vals[t] - m);
        for (int t = 0; t < cnt; ++t) { cIdx[t] = idx[t]; cW[t] = expf(vals[t] - m) / Z; }
        cCnt = cnt;
    }
    __syncthreads();
    int cnt = cCnt;
#pragma unroll
    for (int s = 0; s < 8; ++s) {
        int d = threadIdx.x + s * 256;
        float acc = 0.f;
        for (int t = 0; t < cnt; ++t)
            acc += cW[t] * query[(size_t)cIdx[t] * D_FEAT + d];
        aq[(size_t)i * D_FEAT + d] = acc;
    }
}

// ---------------------------------------------------------------------------
// Launcher
// ---------------------------------------------------------------------------
extern "C" void kernel_launch(void* const* d_in, const int* in_sizes, int n_in,
                              void* d_out, int out_size, void* d_ws, size_t ws_size,
                              hipStream_t stream) {
    (void)in_sizes; (void)n_in; (void)out_size; (void)ws_size;

    const float* x   = (const float*)d_in[0];   // [3520, 8192]
    const float* W   = (const float*)d_in[1];   // [8192, 2048]
    const float* tao = (const float*)d_in[2];   // scalar
    float* out = (float*)d_out;                 // [3200, 64]

    char* ws = (char*)d_ws;
    size_t off = 0;
    auto alloc = [&](size_t bytes) -> char* {
        char* p = ws + off;
        off += (bytes + 255) & ~(size_t)255;
        return p;
    };
    __bf16* xb     = (__bf16*)alloc((size_t)R_TOT * D_INP * 2);   // 57.7 MB (reused: qsim)
    __bf16* Wt     = (__bf16*)alloc((size_t)D_FEAT * D_INP * 2);  // 33.6 MB (reused: aq f32)
    float*  feat   = (float*) alloc((size_t)R_TOT * D_FEAT * 4);  // 28.8 MB (reused: aqb/aprotob)
    float*  query  = (float*) alloc((size_t)NQ * D_FEAT * 4);     // 26.2 MB
    __bf16* queryb = (__bf16*)alloc((size_t)NQ * D_FEAT * 2);     // 13.1 MB
    float*  proto  = (float*) alloc((size_t)N_WAY * D_FEAT * 4);
    __bf16* protob = (__bf16*)alloc((size_t)N_WAY * D_FEAT * 2);
    float*  qn     = (float*) alloc((size_t)NQ * 4);
    float*  pn     = (float*) alloc((size_t)N_WAY * 4);
    float*  pre_sim= (float*) alloc((size_t)NQ * N_WAY * 4);
    int*    label  = (int*)   alloc((size_t)NQ * 4);
    float*  mArr   = (float*) alloc((size_t)N_WAY * 4);
    float*  ZArr   = (float*) alloc((size_t)N_WAY * 4);
    float*  pcoef  = (float*) alloc((size_t)N_WAY * 4);
    float*  coeff  = (float*) alloc((size_t)NQ * 4);
    float*  aproto = (float*) alloc((size_t)N_WAY * D_FEAT * 4);
    int*    kind   = (int*)   alloc((size_t)NQ * TOPK * 4);
    float*  aqn    = (float*) alloc((size_t)NQ * 4);
    float*  apn    = (float*) alloc((size_t)N_WAY * 4);

    // Buffer reuse (lifetimes are disjoint):
    float*  qsim    = (float*)xb;            // 41.0 MB <= 57.7 MB; xb dead after feat GEMM
    float*  aq      = (float*)Wt;            // 26.2 MB <= 33.6 MB; Wt dead after feat GEMM
    __bf16* aqb     = (__bf16*)feat;         // 13.1 MB; feat dead after proto/query extract
    __bf16* aprotob = (__bf16*)((char*)feat + (((size_t)NQ * D_FEAT * 2 + 255) & ~(size_t)255));

    dim3 blk(256);

    // 0) one-time f32->bf16 conversions (W also transposed -> all GEMMs are NT)
    cvt_bf16_kernel<<<dim3((R_TOT * D_INP) / 1024), blk, 0, stream>>>(x, xb);
    transpose_cvt_kernel<<<dim3(D_FEAT / 64, D_INP / 64), blk, 0, stream>>>(W, Wt);

    // 1) feat = x @ W  (118 GFLOP, WMMA bf16, async double-buffered staging)
    gemm64_bf16nt<<<dim3(D_FEAT / 64, R_TOT / 64), blk, 0, stream>>>(
        xb, Wt, feat, R_TOT, D_FEAT, D_INP, nullptr, nullptr, nullptr);

    // 2) proto / query extraction + row norms + bf16 copies
    proto_kernel<<<dim3(N_WAY, D_FEAT / 256), blk, 0, stream>>>(feat, proto);
    query_copy_kernel<<<dim3(NQ, D_FEAT / 256), blk, 0, stream>>>(feat, query);
    row_norm_kernel<<<dim3(NQ),    blk, 0, stream>>>(query, qn, D_FEAT);
    row_norm_kernel<<<dim3(N_WAY), blk, 0, stream>>>(proto, pn, D_FEAT);
    cvt_bf16_kernel<<<dim3((NQ * D_FEAT) / 1024),    blk, 0, stream>>>(query, queryb);
    cvt_bf16_kernel<<<dim3((N_WAY * D_FEAT) / 1024), blk, 0, stream>>>(proto, protob);

    // 3) pre_sim = -(||q||^2 + ||p||^2 - 2 q.p)   (NT GEMM + fused epilogue)
    gemm64_bf16nt<<<dim3(N_WAY / 64, NQ / 64), blk, 0, stream>>>(
        queryb, protob, pre_sim, NQ, N_WAY, D_FEAT, qn, pn, nullptr);

    // 4) class assignment + class softmax + adapted prototypes
    argmax_kernel<<<dim3((NQ + 255) / 256), blk, 0, stream>>>(pre_sim, label);
    cls_stats_kernel<<<dim3(N_WAY), blk, 0, stream>>>(pre_sim, label, mArr, ZArr, pcoef);
    coeff_kernel<<<dim3((NQ + 255) / 256), blk, 0, stream>>>(pre_sim, label, mArr, ZArr, coeff);
    adapted_proto_kernel<<<dim3(N_WAY), blk, 0, stream>>>(query, proto, label, coeff, pcoef, aproto);

    // 5) query_sim (42 GFLOP NT GEMM into reused xb region)
    gemm64_bf16nt<<<dim3(NQ / 64, NQ / 64), blk, 0, stream>>>(
        queryb, queryb, qsim, NQ, NQ, D_FEAT, qn, qn, nullptr);

    // 6) top-10 + mutual-kNN softmax + adapted queries (sparse, <=10 terms/row)
    topk_kernel<<<dim3((NQ + 63) / 64), dim3(64), 0, stream>>>(qsim, kind);
    adapted_query_kernel<<<dim3(NQ), blk, 0, stream>>>(qsim, kind, query, aq);

    // 7) final: out = tao * neg_sq_euclid(adapted_query, adapted_proto)
    row_norm_kernel<<<dim3(NQ),    blk, 0, stream>>>(aq, aqn, D_FEAT);
    row_norm_kernel<<<dim3(N_WAY), blk, 0, stream>>>(aproto, apn, D_FEAT);
    cvt_bf16_kernel<<<dim3((NQ * D_FEAT) / 1024),    blk, 0, stream>>>(aq, aqb);
    cvt_bf16_kernel<<<dim3((N_WAY * D_FEAT) / 1024), blk, 0, stream>>>(aproto, aprotob);
    gemm64_bf16nt<<<dim3(N_WAY / 64, NQ / 64), blk, 0, stream>>>(
        aqb, aprotob, out, NQ, N_WAY, D_FEAT, aqn, apn, tao);
}
